// GDTLayer_77601469104501
// MI455X (gfx1250) — matrix operand order repaired and checked by reading, not compiled, not run on hardware
//
#include <hip/hip_runtime.h>
#include <hip/hip_bf16.h>
#include <math.h>

#define N_NODES 50000
#define DEG 16
#define DIN 128
#define DOUT 128
#define NHEAD 8
#define HDIM 16
#define TOPK 5
#define ALPHA 0.15f
#define SLOPE 0.2f

typedef unsigned short u16;
typedef __attribute__((ext_vector_type(16))) __bf16 v16bf;
typedef __attribute__((ext_vector_type(8)))  float  v8f;

__device__ __forceinline__ u16 f2bf(float f) {
    unsigned int u = __float_as_uint(f);
    unsigned int r = u + 0x7FFFu + ((u >> 16) & 1u);   // round-to-nearest-even
    return (u16)(r >> 16);
}

// ---------------------------------------------------------------------------
// f32 -> bf16 conversion (weights)
// ---------------------------------------------------------------------------
__global__ __launch_bounds__(256) void cvt_bf16_kernel(const float* __restrict__ in,
                                                       u16* __restrict__ out, int n) {
    int i = blockIdx.x * 256 + threadIdx.x;
    if (i < n) out[i] = f2bf(in[i]);
}

// ---------------------------------------------------------------------------
// LayerNorm over 128 dims, one wave32 per node, 4 elems/lane.
// RES=false: y = LN(in)            -> bf16 out
// RES=true : r = in + res; rst=r;  y = LN(r) -> bf16 out
// ---------------------------------------------------------------------------
template<bool RES>
__global__ __launch_bounds__(32) void ln_kernel(const float* __restrict__ in,
                                                const float* __restrict__ res,
                                                const float* __restrict__ g,
                                                const float* __restrict__ b,
                                                float* __restrict__ rst_out,
                                                u16* __restrict__ out_bf) {
    int i = blockIdx.x;
    int lane = threadIdx.x;
    const float* p = in + (size_t)i * DIN + lane * 4;
    float4 v = *(const float4*)p;
    if (RES) {
        float4 r = *(const float4*)(res + (size_t)i * DIN + lane * 4);
        v.x += r.x; v.y += r.y; v.z += r.z; v.w += r.w;
        *(float4*)(rst_out + (size_t)i * DIN + lane * 4) = v;
    }
    float s = v.x + v.y + v.z + v.w;
    #pragma unroll
    for (int o = 16; o > 0; o >>= 1) s += __shfl_xor(s, o, 32);
    float mean = s * (1.0f / DIN);
    float dx = v.x - mean, dy = v.y - mean, dz = v.z - mean, dw = v.w - mean;
    float q = dx * dx + dy * dy + dz * dz + dw * dw;
    #pragma unroll
    for (int o = 16; o > 0; o >>= 1) q += __shfl_xor(q, o, 32);
    float rstd = rsqrtf(q * (1.0f / DIN) + 1e-5f);
    const float4 gg = *(const float4*)(g + lane * 4);
    const float4 bb = *(const float4*)(b + lane * 4);
    ushort4 o4;
    o4.x = f2bf(dx * rstd * gg.x + bb.x);
    o4.y = f2bf(dy * rstd * gg.y + bb.y);
    o4.z = f2bf(dz * rstd * gg.z + bb.z);
    o4.w = f2bf(dw * rstd * gg.w + bb.w);
    *(ushort4*)(out_bf + (size_t)i * DIN + lane * 4) = o4;
}

// ---------------------------------------------------------------------------
// WMMA bf16 GEMM:  C[M x NT] = A[M x KT] * B[NT x KT]^T  (row-major bf16)
// 128-thread blocks = 4 waves; whole B staged once in LDS (shared by the 4
// waves), each wave owns one 16-row strip. Inner loop: A from registers
// (KT==128: preloaded whole row-strip) or streamed (KT==512), B via
// ds_load_b128 from LDS, v_wmma_f32_16x16x32_bf16 accumulate.
// Operand layouts per ISA 7.12.2 (16-bit A 16x32 / B 32x16):
//   A: m=lane&15, khalf=lane>>4, elem e -> k = kc + 16*(e>>3) + 8*khalf + (e&7)
//   B: n=lane&15, khalf=lane>>4, elem e -> k = kc + 16*(e>>3) + 8*khalf + (e&7)
//   C: n=lane&15, m = 8*(lane>>4) + j
// EPI: 0 = store f32; 1 = +bias, relu, store bf16; 2 = +bias +residual, f32
// ---------------------------------------------------------------------------
template<int EPI, int NT, int KT>
__global__ __launch_bounds__(128) void gemm_bf16_kernel(const u16* __restrict__ A,
                                                        const u16* __restrict__ B,
                                                        const float* __restrict__ bias,
                                                        const float* __restrict__ res,
                                                        float* __restrict__ Cf,
                                                        u16* __restrict__ Cb,
                                                        int Mtiles) {
    extern __shared__ u16 sB[];                 // NT*KT bf16 elements
    int t = threadIdx.x;

    // ---- cooperative stage of the full B matrix into LDS (coalesced b128) ----
    constexpr int TOT = NT * KT;
    #pragma unroll 4
    for (int idx = t * 8; idx < TOT; idx += 128 * 8)
        *(uint4*)(sB + idx) = *(const uint4*)(B + idx);
    __syncthreads();

    int wave = t >> 5, lane = t & 31;
    int strip = blockIdx.x * 4 + wave;
    if (strip >= Mtiles) return;
    int row0  = strip * 16;
    int mrow  = lane & 15;
    int khalf = lane >> 4;

    union U { uint4 u[2]; v16bf v; };

    // KT==128: keep the wave's whole A row-strip in registers (32 VGPRs/lane)
    U ua[KT == 128 ? 4 : 1];
    if constexpr (KT == 128) {
        #pragma unroll
        for (int c = 0; c < 4; ++c) {
            const u16* ap = A + (size_t)(row0 + mrow) * KT + c * 32 + 8 * khalf;
            ua[c].u[0] = *(const uint4*)(ap);
            ua[c].u[1] = *(const uint4*)(ap + 16);
        }
    }

    for (int ng = 0; ng < NT; ng += 128) {
        v8f acc[8];
        #pragma unroll
        for (int nt = 0; nt < 8; ++nt) acc[nt] = (v8f){0,0,0,0,0,0,0,0};

        for (int kc = 0; kc < KT; kc += 32) {
            U a_;
            if constexpr (KT == 128) {
                a_ = ua[kc >> 5];
            } else {
                const u16* ap = A + (size_t)(row0 + mrow) * KT + kc + 8 * khalf;
                a_.u[0] = *(const uint4*)(ap);
                a_.u[1] = *(const uint4*)(ap + 16);
            }
            #pragma unroll
            for (int nt = 0; nt < 8; ++nt) {
                U ub;
                const u16* bp = sB + (ng + nt * 16 + mrow) * KT + kc + 8 * khalf;
                ub.u[0] = *(const uint4*)(bp);
                ub.u[1] = *(const uint4*)(bp + 16);
                acc[nt] = __builtin_amdgcn_wmma_f32_16x16x32_bf16(
                    false, a_.v, false, ub.v, (short)0, acc[nt], false, false);
            }
        }
        #pragma unroll
        for (int nt = 0; nt < 8; ++nt) {
            int n = ng + nt * 16 + mrow;
            #pragma unroll
            for (int j = 0; j < 8; ++j) {
                int m = row0 + 8 * khalf + j;
                float v = acc[nt][j];
                if (EPI == 0) {
                    Cf[(size_t)m * NT + n] = v;
                } else if (EPI == 1) {
                    v += bias[n];
                    v = v > 0.0f ? v : 0.0f;
                    Cb[(size_t)m * NT + n] = f2bf(v);
                } else {
                    v += bias[n] + res[(size_t)m * NT + n];
                    Cf[(size_t)m * NT + n] = v;
                }
            }
        }
    }
}

// ---------------------------------------------------------------------------
// Attention: scores, edge-softmax over DEG, top-K mask, renormalize.
// One block (128 thr) per dst node: thread t -> edge e = t>>3, head h = t&7.
// a_n layout: [N][H][DEG]
// ---------------------------------------------------------------------------
__global__ __launch_bounds__(128) void attn_kernel(const float* __restrict__ fh,
                                                   const float* __restrict__ ft,
                                                   const int* __restrict__ src,
                                                   const float* __restrict__ attn,
                                                   float* __restrict__ an) {
    int i = blockIdx.x;
    int t = threadIdx.x;
    int e  = t >> 3;
    int hh = t & 7;
    __shared__ float sc[NHEAD][DEG];
    __shared__ float kth[NHEAD];

    int s = src[(size_t)i * DEG + e];
    const float* fhp = fh + (size_t)s * DOUT + hh * HDIM;
    const float* ftp = ft + (size_t)i * DOUT + hh * HDIM;
    const float* ap  = attn + hh * HDIM;
    float acc = 0.0f;
    #pragma unroll
    for (int d = 0; d < HDIM; ++d) {
        float p = fhp[d] * ftp[d];
        p = p > 0.0f ? p : p * SLOPE;
        acc += p * ap[d];
    }
    acc *= 0.708303336f;   // log(DEG+1)/sqrt(HDIM)
    sc[hh][e] = acc;
    __syncthreads();

    float mx = -1e30f;
    #pragma unroll
    for (int e2 = 0; e2 < DEG; ++e2) mx = fmaxf(mx, sc[hh][e2]);
    float ssum = 0.0f;
    #pragma unroll
    for (int e2 = 0; e2 < DEG; ++e2) ssum += __expf(sc[hh][e2] - mx);
    float a = __expf(acc - mx) / ssum;
    __syncthreads();
    sc[hh][e] = a;
    __syncthreads();

    int gt = 0, ge = 0;
    #pragma unroll
    for (int e2 = 0; e2 < DEG; ++e2) {
        float v = sc[hh][e2];
        gt += (v >  a);
        ge += (v >= a);
    }
    if (gt < TOPK && ge >= TOPK) kth[hh] = a;   // unique value; benign ties
    __syncthreads();

    float masked = (a >= kth[hh]) ? a : 0.0f;
    __syncthreads();
    sc[hh][e] = masked;
    __syncthreads();
    float ts = 0.0f;
    #pragma unroll
    for (int e2 = 0; e2 < DEG; ++e2) ts += sc[hh][e2];
    an[((size_t)i * NHEAD + hh) * DEG + e] = masked / ts;
}

// ---------------------------------------------------------------------------
// One PPR diffusion hop: xn[i,:] = (1-a)*sum_e an[i,h,e]*x[src[i,e],:] + a*fe[i,:]
// One block (128 thr) per node; thread = feature dim; gathers are coalesced
// 512B rows resident in the 192MB L2.
// ---------------------------------------------------------------------------
__global__ __launch_bounds__(128) void hop_kernel(const float* __restrict__ x,
                                                  const float* __restrict__ fe,
                                                  const float* __restrict__ an,
                                                  const int* __restrict__ src,
                                                  float* __restrict__ xn) {
    int i = blockIdx.x;
    int t = threadIdx.x;
    int hh = t >> 4;
    __shared__ int   ss[DEG];
    __shared__ float aw[NHEAD][DEG];
    if (t < DEG) ss[t] = src[(size_t)i * DEG + t];
    aw[t >> 4][t & 15] = an[(size_t)i * (NHEAD * DEG) + t];
    __syncthreads();
    float acc = 0.0f;
    #pragma unroll
    for (int e = 0; e < DEG; ++e)
        acc += aw[hh][e] * x[(size_t)ss[e] * DOUT + t];
    xn[(size_t)i * DOUT + t] = (1.0f - ALPHA) * acc + ALPHA * fe[(size_t)i * DOUT + t];
}

// ---------------------------------------------------------------------------
// Host-side launcher
// ---------------------------------------------------------------------------
extern "C" void kernel_launch(void* const* d_in, const int* in_sizes, int n_in,
                              void* d_out, int out_size, void* d_ws, size_t ws_size,
                              hipStream_t stream) {
    const float* feat  = (const float*)d_in[0];
    const int*   src   = (const int*)  d_in[1];
    const float* Wh    = (const float*)d_in[2];
    const float* Wt    = (const float*)d_in[3];
    const float* We    = (const float*)d_in[4];
    const float* attn  = (const float*)d_in[5];
    const float* ln1g  = (const float*)d_in[6];
    const float* ln1b  = (const float*)d_in[7];
    const float* ln2g  = (const float*)d_in[8];
    const float* ln2b  = (const float*)d_in[9];
    const float* w1    = (const float*)d_in[10];
    const float* b1    = (const float*)d_in[11];
    const float* w2    = (const float*)d_in[12];
    const float* b2    = (const float*)d_in[13];
    float* out = (float*)d_out;

    // --- workspace carving (regions reused across phases) ---
    char* ws = (char*)d_ws;
    size_t off = 0;
    auto alloc = [&](size_t bytes) { char* p = ws + off; off += (bytes + 255) & ~(size_t)255; return p; };
    const size_t NF = (size_t)N_NODES * DIN;      // 6.4M elems
    u16*   hbf = (u16*)  alloc(NF * 2);
    float* fh  = (float*)alloc(NF * 4);           // later reused as rst
    float* ft  = (float*)alloc(NF * 4);           // later reused as x ping buffer B
    float* fe  = (float*)alloc(NF * 4);
    float* an  = (float*)alloc(NF * 4);           // DEG*H == 128
    float* xA  = (float*)alloc(NF * 4);
    u16*   ybf = (u16*)  alloc(NF * 2);
    u16*   zbf = (u16*)  alloc((size_t)N_NODES * 512 * 2);
    u16*   whb = (u16*)  alloc(DOUT * DIN * 2);
    u16*   wtb = (u16*)  alloc(DOUT * DIN * 2);
    u16*   web = (u16*)  alloc(DOUT * DIN * 2);
    u16*   w1b = (u16*)  alloc(512 * DOUT * 2);
    u16*   w2b = (u16*)  alloc(DOUT * 512 * 2);
    float* xB  = ft;                               // reuse ftl region for ping-pong
    float* rst = fh;                               // reuse fh region for residual

    const int MTILES = N_NODES / 16;               // 3125
    const int GBLKS  = (MTILES + 3) / 4;           // 4 strips (waves) per block

    // 1) weight conversions
    cvt_bf16_kernel<<<(DOUT * DIN + 255) / 256, 256, 0, stream>>>(Wh, whb, DOUT * DIN);
    cvt_bf16_kernel<<<(DOUT * DIN + 255) / 256, 256, 0, stream>>>(Wt, wtb, DOUT * DIN);
    cvt_bf16_kernel<<<(DOUT * DIN + 255) / 256, 256, 0, stream>>>(We, web, DOUT * DIN);
    cvt_bf16_kernel<<<(512 * DOUT + 255) / 256, 256, 0, stream>>>(w1, w1b, 512 * DOUT);
    cvt_bf16_kernel<<<(DOUT * 512 + 255) / 256, 256, 0, stream>>>(w2, w2b, DOUT * 512);

    // 2) LN1 -> bf16 h
    ln_kernel<false><<<N_NODES, 32, 0, stream>>>(feat, nullptr, ln1g, ln1b, nullptr, hbf);

    // 3) three projections via WMMA (B staged in 32KB LDS per block)
    gemm_bf16_kernel<0, 128, 128><<<GBLKS, 128, 128 * 128 * 2, stream>>>(hbf, whb, nullptr, nullptr, fh, nullptr, MTILES);
    gemm_bf16_kernel<0, 128, 128><<<GBLKS, 128, 128 * 128 * 2, stream>>>(hbf, wtb, nullptr, nullptr, ft, nullptr, MTILES);
    gemm_bf16_kernel<0, 128, 128><<<GBLKS, 128, 128 * 128 * 2, stream>>>(hbf, web, nullptr, nullptr, fe, nullptr, MTILES);

    // 4) attention + softmax + top-k renorm
    attn_kernel<<<N_NODES, 128, 0, stream>>>(fh, ft, src, attn, an);

    // 5) 5 PPR hops (x0 = fe), ping-pong ending in xA
    hop_kernel<<<N_NODES, 128, 0, stream>>>(fe, fe, an, src, xA);
    hop_kernel<<<N_NODES, 128, 0, stream>>>(xA, fe, an, src, xB);
    hop_kernel<<<N_NODES, 128, 0, stream>>>(xB, fe, an, src, xA);
    hop_kernel<<<N_NODES, 128, 0, stream>>>(xA, fe, an, src, xB);
    hop_kernel<<<N_NODES, 128, 0, stream>>>(xB, fe, an, src, xA);

    // 6) rst = x + feat ; y = LN2(rst) -> bf16
    ln_kernel<true><<<N_NODES, 32, 0, stream>>>(xA, feat, ln2g, ln2b, rst, ybf);

    // 7) FFN: z = relu(y @ w1^T + b1) (bf16, 128KB LDS), out = z @ w2^T + b2 + rst
    gemm_bf16_kernel<1, 512, 128><<<GBLKS, 128, 512 * 128 * 2, stream>>>(ybf, w1b, b1, nullptr, nullptr, zbf, MTILES);
    gemm_bf16_kernel<2, 128, 512><<<GBLKS, 128, 128 * 512 * 2, stream>>>(zbf, w2b, b2, rst, out, nullptr, MTILES);
}
